// sub_MGCANet_84774064489099
// MI455X (gfx1250) — matrix operand order repaired and checked by reading, not compiled, not run on hardware
//
#include <hip/hip_runtime.h>
#include <hip/hip_bf16.h>
#include <math.h>

// ---------------------------------------------------------------------------
// MGCANet forward for MI455X (gfx1250), WMMA-centric.
//  - All dense GEMMs via v_wmma_f32_16x16x32_f16 (f16 operands, fp32 accum).
//  - gemm_tn (attention Gram): software-pipelined contiguous b128 fragments.
//  - gemm_nt: X tile staged into LDS; uses GLOBAL_LOAD_ASYNC_TO_LDS_B128 +
//    s_wait_asynccnt when the toolchain exposes the builtins, else manual
//    transposed staging (ds_load_b128 fragments).
//  - N=2000 padded to NP=2048; pad columns of every f16 operand zeroed.
// ---------------------------------------------------------------------------

#define Bsz 16
#define Nn  2000
#define NP  2048
#define Cc  128
#define Kk  9
#define ICc 32

#if defined(__has_builtin)
#  if __has_builtin(__builtin_amdgcn_global_load_async_to_lds_b128) && \
      __has_builtin(__builtin_amdgcn_s_wait_asynccnt)
#    define HAVE_ASYNC_LDS 1
#  endif
#endif
#ifndef HAVE_ASYNC_LDS
#  define HAVE_ASYNC_LDS 0
#endif

typedef __attribute__((ext_vector_type(16))) _Float16 v16h;
typedef __attribute__((ext_vector_type(8)))  _Float16 v8h;
typedef __attribute__((ext_vector_type(8)))  float    v8f;
typedef int v4i __attribute__((vector_size(16)));   // 128-bit payload for async LDS DMA

__device__ __forceinline__ v16h cat8(v8h lo, v8h hi) {
    return __builtin_shufflevector(lo, hi, 0,1,2,3,4,5,6,7,8,9,10,11,12,13,14,15);
}
__device__ __forceinline__ float gelu_f(float x) {
    return 0.5f * x * (1.0f + erff(x * 0.70710678118654752f));
}
__device__ __forceinline__ float sigmoid_f(float x) {
    return 1.0f / (1.0f + __expf(-x));
}

// ---------------------------------------------------------------------------
// gemm_nt: Y[b] = act(A[b?] * X[b] + bias + addv) * postScale (+ add1 + add2)
//   A16: f16 [M][Kd] (aStride=0 -> shared weights, else per-batch).
//   X:   f16 [b][Kd][ldX]; 32x64 tile staged into LDS per K-step.
// Block (32,4): 4 waves = 4 adjacent n-tiles (64 cols); grid.x = NP/64.
// ---------------------------------------------------------------------------
template<int ACT>
__global__ void gemm_nt_kernel(
    const _Float16* __restrict__ A16, long long aStride,
    const _Float16* __restrict__ X, long long xStride, int ldX,
    const float* __restrict__ bias,
    const float* __restrict__ addv,   // [B][M]
    const float* __restrict__ add1,   // [B][M][Nvalid]
    const float* __restrict__ add2,   // [B][M][Nvalid]
    float* __restrict__ Y32,          // [B][M][Nvalid]
    _Float16* __restrict__ Y16, int ldY16,
    int M, int Kd, int Nvalid, float postScale)
{
#if HAVE_ASYNC_LDS
    __shared__ _Float16 Xs[32][64];               // row-major [k][col], 4KB
#else
    __shared__ _Float16 XsT[64 * 32];             // [colLocal][k], 4KB
#endif
    const int lane = threadIdx.x;                 // wave32 lane
    const int tloc = threadIdx.y;                 // n-tile within block
    const int tid  = tloc * 32 + lane;
    const int g    = lane >> 4;
    const int h16  = lane & 15;
    const int nb   = blockIdx.x * 64;
    const int m0   = blockIdx.y * 16;
    const int b    = blockIdx.z;

    const _Float16* Ab = A16 + (long long)b * aStride;
    const _Float16* Xb = X   + (long long)b * xStride;
    const int arow = m0 + h16;

    v8f acc = {};
    for (int k0 = 0; k0 < Kd; k0 += 32) {
#if HAVE_ASYNC_LDS
        // Async DMA the 32x64 (f16) tile into LDS: 256 b128 ops, 2 per thread.
#pragma unroll
        for (int o = 0; o < 2; ++o) {
            int idx = tid * 2 + o;
            int row = idx >> 3, seg = (idx & 7) * 8;
            const _Float16* gsrc = Xb + (long long)(k0 + row) * ldX + nb + seg;
            __builtin_amdgcn_global_load_async_to_lds_b128(
                (v4i*)gsrc, (v4i*)&Xs[row][seg], 0, 0);
        }
        __builtin_amdgcn_s_wait_asynccnt(0);
        __syncthreads();
#else
        if (k0 + 32 < Kd && tid < 32)             // hint next X block into cache
            __builtin_prefetch((const void*)(Xb + (long long)(k0 + 32 + tid) * ldX + nb), 0, 3);
        // stage X[k0..k0+31][nb..nb+63] transposed (coalesced global reads)
        for (int i = tid; i < 32 * 64; i += 128) {
            int kr = i >> 6, cl = i & 63;
            XsT[cl * 32 + kr] = Xb[(long long)(k0 + kr) * ldX + nb + cl];
        }
        __syncthreads();
#endif
        // A fragment: lane holds row m0+h16; K in [k0+16g,+16)
        const _Float16* ap = Ab + (long long)arow * Kd + k0;
        v8h alo = *(const v8h*)(ap + 16 * g);
        v8h ahi = *(const v8h*)(ap + 16 * g + 8);
        // B fragment: lane holds col n; K = 16g + h
#if HAVE_ASYNC_LDS
        v16h bv;
#pragma unroll
        for (int h = 0; h < 16; ++h) bv[h] = Xs[16 * g + h][tloc * 16 + h16];
        acc = __builtin_amdgcn_wmma_f32_16x16x32_f16(false, cat8(alo, ahi),
                                                     false, bv,
                                                     (short)0, acc, false, false);
#else
        const _Float16* bp = &XsT[(tloc * 16 + h16) * 32 + 16 * g];
        v8h blo = *(const v8h*)(bp);
        v8h bhi = *(const v8h*)(bp + 8);
        acc = __builtin_amdgcn_wmma_f32_16x16x32_f16(false, cat8(alo, ahi),
                                                     false, cat8(blo, bhi),
                                                     (short)0, acc, false, false);
#endif
        __syncthreads();
    }
#pragma unroll
    for (int r = 0; r < 8; ++r) {
        const int m = m0 + r + 8 * g;
        const int n = nb + tloc * 16 + h16;
        float v = acc[r];
        if (bias) v += bias[m];
        if (addv) v += addv[b * M + m];
        if      (ACT == 1) v = fmaxf(v, 0.0f);
        else if (ACT == 2) v = gelu_f(v);
        else if (ACT == 3) v = sigmoid_f(v);
        v *= postScale;
        const bool inN = (n < Nvalid);
        if (inN) {
            long long fi = ((long long)b * M + m) * Nvalid + n;
            if (add1) v += add1[fi];
            if (add2) v += add2[fi];
            if (Y32)  Y32[fi] = v;
        }
        if (Y16) Y16[((long long)b * M + m) * ldY16 + n] = inN ? (_Float16)v : (_Float16)0.0f;
    }
}

// ---------------------------------------------------------------------------
// gemm_tn: Y[b][m][n] = sum_k A[b][m][k] * X[b][n][k]   (attention Gram)
// Fragments are contiguous b128 pairs; software-pipelined (next-step loads
// issue before this step's wmma so load latency overlaps the matrix op).
// ---------------------------------------------------------------------------
__global__ void gemm_tn_kernel(
    const _Float16* __restrict__ A, long long aStride,
    const _Float16* __restrict__ X, long long xStride, int ld,
    float* __restrict__ Y32, int M, int Kd, int Ncols)
{
    const int lane = threadIdx.x;
    const int g    = lane >> 4;
    const int h16  = lane & 15;
    const int n0   = (blockIdx.x * blockDim.y + threadIdx.y) * 16;
    const int m0   = blockIdx.y * 16;
    const int b    = blockIdx.z;
    const _Float16* Ab = A + (long long)b * aStride + (long long)(m0 + h16) * ld;
    const _Float16* Xb = X + (long long)b * xStride + (long long)(n0 + h16) * ld;

    v8f acc = {};
    v8h a0 = *(const v8h*)(Ab + 16 * g);
    v8h a1 = *(const v8h*)(Ab + 16 * g + 8);
    v8h b0 = *(const v8h*)(Xb + 16 * g);
    v8h b1 = *(const v8h*)(Xb + 16 * g + 8);
    for (int k0 = 0; k0 < Kd; k0 += 32) {
        const int kn = (k0 + 32 < Kd) ? (k0 + 32) : k0;   // branchless clamp
        v8h na0 = *(const v8h*)(Ab + kn + 16 * g);
        v8h na1 = *(const v8h*)(Ab + kn + 16 * g + 8);
        v8h nb0 = *(const v8h*)(Xb + kn + 16 * g);
        v8h nb1 = *(const v8h*)(Xb + kn + 16 * g + 8);
        __builtin_prefetch((const void*)(Xb + kn + 64), 0, 3);
        acc = __builtin_amdgcn_wmma_f32_16x16x32_f16(false, cat8(a0, a1),
                                                     false, cat8(b0, b1),
                                                     (short)0, acc, false, false);
        a0 = na0; a1 = na1; b0 = nb0; b1 = nb1;
    }
#pragma unroll
    for (int r = 0; r < 8; ++r)
        Y32[((long long)b * M + m0 + r + 8 * g) * Ncols + n0 + h16] = acc[r];
}

// f32 -> f16 weight conversion
__global__ void cvt16_kernel(const float* __restrict__ src, _Float16* __restrict__ dst, int n)
{
    int i = blockIdx.x * 256 + threadIdx.x;
    if (i < n) dst[i] = (_Float16)src[i];
}

// ---------------------------------------------------------------------------
// KNN (top-9 by -||d||^2) + graph feature + conv1 + relu + max over k.
// Entire batch's points staged in LDS (2000*4 floats = 32KB of 320KB).
// ---------------------------------------------------------------------------
__global__ void knn_conv1_kernel(const float* __restrict__ x,      // [B][N][4]
                                 const float* __restrict__ W,      // [128][8]
                                 const float* __restrict__ bias,   // [128]
                                 float* __restrict__ f32o,         // [B][128][N]
                                 _Float16* __restrict__ f16o)      // [B][128][NP]
{
    __shared__ float pts[Nn * 4];
    __shared__ float Ws[Cc * 8];
    __shared__ float Bs[Cc];
    const int b   = blockIdx.y;
    const int tid = threadIdx.x;
    for (int i = tid; i < Nn * 4; i += 256) pts[i] = x[(long long)b * Nn * 4 + i];
    for (int i = tid; i < Cc * 8; i += 256) Ws[i] = W[i];
    for (int i = tid; i < Cc;     i += 256) Bs[i] = bias[i];
    __syncthreads();

    const int n = blockIdx.x * 256 + tid;
    if (n >= NP) return;
    if (n >= Nn) {                                 // zero f16 pad columns
        for (int o = 0; o < Cc; ++o)
            f16o[((long long)b * Cc + o) * NP + n] = (_Float16)0.0f;
        return;
    }
    const float px = pts[n*4+0], py = pts[n*4+1], pz = pts[n*4+2], pw = pts[n*4+3];

    float bd[Kk]; int bj[Kk];
#pragma unroll
    for (int t = 0; t < Kk; ++t) { bd[t] = -1e30f; bj[t] = 0; }
    for (int j = 0; j < Nn; ++j) {
        float dx = px - pts[j*4+0], dy = py - pts[j*4+1];
        float dz = pz - pts[j*4+2], dw = pw - pts[j*4+3];
        float d = -(dx*dx + dy*dy + dz*dz + dw*dw);
        if (d > bd[Kk-1]) {
            int t = Kk - 1;
            while (t > 0 && bd[t-1] < d) { bd[t] = bd[t-1]; bj[t] = bj[t-1]; --t; }
            bd[t] = d; bj[t] = j;
        }
    }
    float dif[Kk][4];
#pragma unroll
    for (int t = 0; t < Kk; ++t) {
        int j = bj[t];
        dif[t][0] = px - pts[j*4+0]; dif[t][1] = py - pts[j*4+1];
        dif[t][2] = pz - pts[j*4+2]; dif[t][3] = pw - pts[j*4+3];
    }
    for (int o = 0; o < Cc; ++o) {
        const float* w = &Ws[o * 8];
        float base = Bs[o] + w[0]*px + w[1]*py + w[2]*pz + w[3]*pw;
        float mh = -1e30f;
#pragma unroll
        for (int t = 0; t < Kk; ++t) {
            float hh = base + w[4]*dif[t][0] + w[5]*dif[t][1]
                            + w[6]*dif[t][2] + w[7]*dif[t][3];
            mh = fmaxf(mh, hh);
        }
        float val = fmaxf(mh, 0.0f);   // max(relu(h)) == relu(max(h))
        f32o[((long long)b * Cc + o) * Nn + n] = val;
        f16o[((long long)b * Cc + o) * NP + n] = (_Float16)val;
    }
}

// gc = g1_W*x[0:2] + g1_b + g2_W*x[2:4] + g2_b  -> f16 [B][C][NP], pads zero
__global__ void gc_kernel(const float* __restrict__ x,
                          const float* __restrict__ g1W, const float* __restrict__ g1b,
                          const float* __restrict__ g2W, const float* __restrict__ g2b,
                          _Float16* __restrict__ gc16)
{
    const int b = blockIdx.y;
    const int n = blockIdx.x * 256 + threadIdx.x;
    if (n >= NP) return;
    if (n >= Nn) {
        for (int c = 0; c < Cc; ++c)
            gc16[((long long)b * Cc + c) * NP + n] = (_Float16)0.0f;
        return;
    }
    const float* xp = x + ((long long)b * Nn + n) * 4;
    float x0 = xp[0], x1 = xp[1], x2 = xp[2], x3 = xp[3];
    for (int c = 0; c < Cc; ++c) {
        float v = g1W[c*2+0]*x0 + g1W[c*2+1]*x1 + g1b[c]
                + g2W[c*2+0]*x2 + g2W[c*2+1]*x3 + g2b[c];
        gc16[((long long)b * Cc + c) * NP + n] = (_Float16)v;
    }
}

__global__ void add16_kernel(_Float16* __restrict__ dst, const _Float16* __restrict__ src, int n)
{
    int i = blockIdx.x * 256 + threadIdx.x;
    if (i < n) dst[i] = (_Float16)((float)dst[i] + (float)src[i]);
}

// softmax over d (128) per (b,c) row; writes f16 for the attn*v WMMA GEMM
__global__ void softmax_kernel(const float* __restrict__ a32, _Float16* __restrict__ a16)
{
    const int row = blockIdx.x;      // b*128 + c
    const int d   = threadIdx.x;     // 128
    __shared__ float sm[128];
    float v = a32[(long long)row * Cc + d];
    sm[d] = v; __syncthreads();
    for (int s = 64; s > 0; s >>= 1) { if (d < s) sm[d] = fmaxf(sm[d], sm[d+s]); __syncthreads(); }
    float mx = sm[0]; __syncthreads();
    float e = __expf(v - mx);
    sm[d] = e; __syncthreads();
    for (int s = 64; s > 0; s >>= 1) { if (d < s) sm[d] += sm[d+s]; __syncthreads(); }
    a16[(long long)row * Cc + d] = (_Float16)(e / sm[0]);
}

// LayerNorm over C per point -> f32 + f16 (pads zero)
__global__ void ln_kernel(const float* __restrict__ f1, const float* __restrict__ g,
                          const float* __restrict__ bta,
                          float* __restrict__ o32, _Float16* __restrict__ o16)
{
    const int b = blockIdx.y;
    const int n = blockIdx.x * 256 + threadIdx.x;
    if (n >= NP) return;
    if (n >= Nn) {
        for (int c = 0; c < Cc; ++c)
            o16[((long long)b * Cc + c) * NP + n] = (_Float16)0.0f;
        return;
    }
    float s = 0.f, ss = 0.f;
    for (int c = 0; c < Cc; ++c) {
        float v = f1[((long long)b * Cc + c) * Nn + n];
        s += v; ss += v * v;
    }
    float mu = s / Cc;
    float inv = rsqrtf(ss / Cc - mu * mu + 1e-6f);
    for (int c = 0; c < Cc; ++c) {
        long long i = ((long long)b * Cc + c) * Nn + n;
        float v = (f1[i] - mu) * inv * g[c] + bta[c];
        o32[i] = v;
        o16[((long long)b * Cc + c) * NP + n] = (_Float16)v;
    }
}

// avg & max over n per (b,c)
__global__ void pool_kernel(const float* __restrict__ ic,
                            float* __restrict__ avg, float* __restrict__ mx)
{
    const int c = blockIdx.x, b = blockIdx.y, tid = threadIdx.x;
    __shared__ float ss[256], sm[256];
    float s = 0.f, m = -1e30f;
    for (int n = tid; n < Nn; n += 256) {
        float v = ic[((long long)b * Cc + c) * Nn + n];
        s += v; m = fmaxf(m, v);
    }
    ss[tid] = s; sm[tid] = m; __syncthreads();
    for (int st = 128; st > 0; st >>= 1) {
        if (tid < st) { ss[tid] += ss[tid+st]; sm[tid] = fmaxf(sm[tid], sm[tid+st]); }
        __syncthreads();
    }
    if (tid == 0) { avg[b * Cc + c] = ss[0] / (float)Nn; mx[b * Cc + c] = sm[0]; }
}

// s23[b][c] = ga2(gelu(ga1*avg+b))+b + gm2(gelu(gm1*mx+b))+b
__global__ void s23_kernel(const float* __restrict__ avg, const float* __restrict__ mx,
                           const float* __restrict__ a1W, const float* __restrict__ a1b,
                           const float* __restrict__ a2W, const float* __restrict__ a2b,
                           const float* __restrict__ m1W, const float* __restrict__ m1b,
                           const float* __restrict__ m2W, const float* __restrict__ m2b,
                           float* __restrict__ s23)
{
    const int b = blockIdx.x, t = threadIdx.x;   // 128 threads
    __shared__ float ha[ICc], hm[ICc];
    if (t < ICc) {
        float sa = a1b[t], sm_ = m1b[t];
        for (int c = 0; c < Cc; ++c) {
            sa  += a1W[t * Cc + c] * avg[b * Cc + c];
            sm_ += m1W[t * Cc + c] * mx[b * Cc + c];
        }
        ha[t] = gelu_f(sa); hm[t] = gelu_f(sm_);
    }
    __syncthreads();
    float v = a2b[t] + m2b[t];
    for (int i = 0; i < ICc; ++i)
        v += a2W[t * ICc + i] * ha[i] + m2W[t * ICc + i] * hm[i];
    s23[b * Cc + t] = v;
}

__global__ void zero_kernel(float* __restrict__ p, int n)
{
    int i = blockIdx.x * blockDim.x + threadIdx.x;
    if (i < n) p[i] = 0.0f;
}

// LN2 over C, feat=ln1+ln2, logits=fin_W*feat+fin_b, w=exp(l1)*sigmoid(l0)
__global__ void ln2_w_kernel(const float* __restrict__ mb, const float* __restrict__ fln1,
                             const float* __restrict__ g, const float* __restrict__ bta,
                             const float* __restrict__ finW, const float* __restrict__ finb,
                             float* __restrict__ wbuf, float* __restrict__ wsum)
{
    const int b = blockIdx.y;
    const int n = blockIdx.x * 256 + threadIdx.x;
    float wv = 0.0f;
    if (n < Nn) {
        float s = 0.f, ss = 0.f;
        for (int c = 0; c < Cc; ++c) {
            float v = mb[((long long)b * Cc + c) * Nn + n];
            s += v; ss += v * v;
        }
        float mu = s / Cc;
        float inv = rsqrtf(ss / Cc - mu * mu + 1e-6f);
        float l0 = finb[0], l1 = finb[1];
        for (int c = 0; c < Cc; ++c) {
            long long i = ((long long)b * Cc + c) * Nn + n;
            float fo = (mb[i] - mu) * inv * g[c] + bta[c] + fln1[i];
            l0 += finW[c] * fo;
            l1 += finW[Cc + c] * fo;
        }
        wv = __expf(l1) * sigmoid_f(l0);
        wbuf[(long long)b * Nn + n] = wv;
    }
    __shared__ float red[256];
    red[threadIdx.x] = wv; __syncthreads();
    for (int st = 128; st > 0; st >>= 1) {
        if ((int)threadIdx.x < st) red[threadIdx.x] += red[threadIdx.x + st];
        __syncthreads();
    }
    if (threadIdx.x == 0) atomicAdd(&wsum[b], red[0]);
}

// XwX[b] = sum_n w_n * X_n X_n^T  (9x9, w normalized by wsum+1e-5)
__global__ void xwx_kernel(const float* __restrict__ x, const float* __restrict__ wbuf,
                           const float* __restrict__ wsum, float* __restrict__ XwX)
{
    __shared__ float acc[81];
    const int b = blockIdx.x, tid = threadIdx.x;
    for (int i = tid; i < 81; i += 256) acc[i] = 0.0f;
    __syncthreads();
    float loc[45];
#pragma unroll
    for (int i = 0; i < 45; ++i) loc[i] = 0.0f;
    const float inv = 1.0f / (wsum[b] + 1e-5f);
    for (int n = tid; n < Nn; n += 256) {
        const float* xp = x + ((long long)b * Nn + n) * 4;
        float x0 = xp[0], x1 = xp[1], x2 = xp[2], x3 = xp[3];
        float Xv[9] = { x2*x0, x2*x1, x2, x3*x0, x3*x1, x3, x0, x1, 1.0f };
        float w = wbuf[(long long)b * Nn + n] * inv;
        int t = 0;
#pragma unroll
        for (int i = 0; i < 9; ++i) {
            float wxi = w * Xv[i];
#pragma unroll
            for (int j = i; j < 9; ++j) loc[t++] += wxi * Xv[j];
        }
    }
    {
        int t = 0;
#pragma unroll
        for (int i = 0; i < 9; ++i)
#pragma unroll
            for (int j = i; j < 9; ++j) atomicAdd(&acc[i * 9 + j], loc[t++]);
    }
    __syncthreads();
    for (int idx = tid; idx < 81; idx += 256) {
        int i = idx / 9, j = idx % 9;
        XwX[b * 81 + idx] = (i <= j) ? acc[i * 9 + j] : acc[j * 9 + i];
    }
}

// Cyclic Jacobi eigh of 9x9 per batch; output eigenvector of smallest eigenvalue
__global__ void eigh_kernel(const float* __restrict__ XwX, float* __restrict__ out)
{
    const int b = threadIdx.x;
    if (b >= Bsz) return;
    float A[9][9], V[9][9];
    for (int i = 0; i < 9; ++i)
        for (int j = 0; j < 9; ++j) {
            A[i][j] = XwX[b * 81 + i * 9 + j];
            V[i][j] = (i == j) ? 1.0f : 0.0f;
        }
    for (int sweep = 0; sweep < 15; ++sweep) {
        for (int p = 0; p < 8; ++p) {
            for (int q = p + 1; q < 9; ++q) {
                float apq = A[p][q];
                if (fabsf(apq) < 1e-13f) continue;
                float tau = (A[q][q] - A[p][p]) / (2.0f * apq);
                float t = ((tau >= 0.f) ? 1.f : -1.f) / (fabsf(tau) + sqrtf(1.0f + tau * tau));
                float c = rsqrtf(1.0f + t * t);
                float s = t * c;
                for (int k = 0; k < 9; ++k) {
                    float akp = A[k][p], akq = A[k][q];
                    A[k][p] = c * akp - s * akq;
                    A[k][q] = s * akp + c * akq;
                }
                for (int k = 0; k < 9; ++k) {
                    float apk = A[p][k], aqk = A[q][k];
                    A[p][k] = c * apk - s * aqk;
                    A[q][k] = s * apk + c * aqk;
                }
                for (int k = 0; k < 9; ++k) {
                    float vkp = V[k][p], vkq = V[k][q];
                    V[k][p] = c * vkp - s * vkq;
                    V[k][q] = s * vkp + c * vkq;
                }
            }
        }
    }
    int mi = 0; float mv = A[0][0];
    for (int i = 1; i < 9; ++i) if (A[i][i] < mv) { mv = A[i][i]; mi = i; }
    float nrm = 0.f;
    for (int i = 0; i < 9; ++i) nrm += V[i][mi] * V[i][mi];
    nrm = rsqrtf(nrm + 1e-30f);
    for (int i = 0; i < 9; ++i) out[b * 9 + i] = V[i][mi] * nrm;
}

// ---------------------------------------------------------------------------
extern "C" void kernel_launch(void* const* d_in, const int* in_sizes, int n_in,
                              void* d_out, int out_size, void* d_ws, size_t ws_size,
                              hipStream_t stream)
{
    (void)in_sizes; (void)n_in; (void)out_size; (void)ws_size;
    const float* x      = (const float*)d_in[0];
    const float* conv1W = (const float*)d_in[1];
    const float* conv1b = (const float*)d_in[2];
    const float* qW   = (const float*)d_in[3];
    const float* kW   = (const float*)d_in[4];
    const float* vW   = (const float*)d_in[5];
    const float* g1W  = (const float*)d_in[6];
    const float* g1b  = (const float*)d_in[7];
    const float* g2W  = (const float*)d_in[8];
    const float* g2b  = (const float*)d_in[9];
    const float* ln1g = (const float*)d_in[10];
    const float* ln1b = (const float*)d_in[11];
    const float* inW  = (const float*)d_in[12];
    const float* inb  = (const float*)d_in[13];
    const float* l1W  = (const float*)d_in[14];
    const float* l1b  = (const float*)d_in[15];
    const float* l2W  = (const float*)d_in[16];
    const float* l2b  = (const float*)d_in[17];
    const float* ga1W = (const float*)d_in[18];
    const float* ga1b = (const float*)d_in[19];
    const float* ga2W = (const float*)d_in[20];
    const float* ga2b = (const float*)d_in[21];
    const float* gm1W = (const float*)d_in[22];
    const float* gm1b = (const float*)d_in[23];
    const float* gm2W = (const float*)d_in[24];
    const float* gm2b = (const float*)d_in[25];
    const float* outW = (const float*)d_in[26];
    const float* outb = (const float*)d_in[27];
    const float* ln2g = (const float*)d_in[28];
    const float* ln2b = (const float*)d_in[29];
    const float* finW = (const float*)d_in[30];
    const float* finb = (const float*)d_in[31];

    char* base = (char*)d_ws;
    size_t off = 0;
    auto alloc = [&](size_t bytes) -> void* {
        void* p = base + off;
        off += (bytes + 255) & ~(size_t)255;
        return p;
    };
    const size_t F32B = (size_t)Bsz * Cc * Nn * sizeof(float);
    const size_t F16B = (size_t)Bsz * Cc * NP * sizeof(_Float16);

    float*    fbuf   = (float*)   alloc(F32B);  // f -> f1(cpt+f) -> mb (reused)
    _Float16* f16    = (_Float16*)alloc(F16B);
    _Float16* qs16   = (_Float16*)alloc(F16B);  // relu(q)/sqrt(C)
    _Float16* kg16   = (_Float16*)alloc(F16B);  // relu(k), then k+gc
    _Float16* v16b   = (_Float16*)alloc(F16B);
    _Float16* gc16   = (_Float16*)alloc(F16B);
    float*    attn32 = (float*)   alloc((size_t)Bsz * Cc * Cc * sizeof(float));
    _Float16* attn16 = (_Float16*)alloc((size_t)Bsz * Cc * Cc * sizeof(_Float16));
    float*    fln1   = (float*)   alloc(F32B);
    _Float16* fln116 = (_Float16*)alloc(F16B);
    float*    ic32   = (float*)   alloc(F32B);
    _Float16* ic16   = (_Float16*)alloc(F16B);
    _Float16* hid16  = (_Float16*)alloc((size_t)Bsz * ICc * NP * sizeof(_Float16));
    _Float16* sc16   = (_Float16*)alloc(F16B);
    // f16 weight copies
    _Float16* qW16   = (_Float16*)alloc((size_t)Cc * Cc * sizeof(_Float16));
    _Float16* kW16   = (_Float16*)alloc((size_t)Cc * Cc * sizeof(_Float16));
    _Float16* vW16   = (_Float16*)alloc((size_t)Cc * Cc * sizeof(_Float16));
    _Float16* inW16  = (_Float16*)alloc((size_t)Cc * Cc * sizeof(_Float16));
    _Float16* outW16 = (_Float16*)alloc((size_t)Cc * Cc * sizeof(_Float16));
    _Float16* l1W16  = (_Float16*)alloc((size_t)ICc * Cc * sizeof(_Float16));
    _Float16* l2W16  = (_Float16*)alloc((size_t)Cc * ICc * sizeof(_Float16));
    float*    avgb   = (float*)   alloc((size_t)Bsz * Cc * sizeof(float));
    float*    mxb    = (float*)   alloc((size_t)Bsz * Cc * sizeof(float));
    float*    s23    = (float*)   alloc((size_t)Bsz * Cc * sizeof(float));
    float*    wbuf   = (float*)   alloc((size_t)Bsz * Nn * sizeof(float));
    float*    wsum   = (float*)   alloc((size_t)Bsz * sizeof(float));
    float*    xwx    = (float*)   alloc((size_t)Bsz * 81 * sizeof(float));

    const dim3 blk256(256);
    const dim3 blkg(32, 4);
    const dim3 gridPts(NP / 256, Bsz);
    const long long sF = (long long)Cc * NP;           // per-batch f16 stride
    const dim3 gridG(NP / 64, Cc / 16, Bsz);           // C-row GEMMs over NP cols
    const dim3 gridH(NP / 64, ICc / 16, Bsz);          // IC-row GEMM
    const dim3 gridA(2, Cc / 16, Bsz);                 // 128x128 attention tiles

    // Stage 0: weights -> f16
    cvt16_kernel<<<(Cc*Cc + 255)/256, blk256, 0, stream>>>(qW,   qW16,   Cc*Cc);
    cvt16_kernel<<<(Cc*Cc + 255)/256, blk256, 0, stream>>>(kW,   kW16,   Cc*Cc);
    cvt16_kernel<<<(Cc*Cc + 255)/256, blk256, 0, stream>>>(vW,   vW16,   Cc*Cc);
    cvt16_kernel<<<(Cc*Cc + 255)/256, blk256, 0, stream>>>(inW,  inW16,  Cc*Cc);
    cvt16_kernel<<<(Cc*Cc + 255)/256, blk256, 0, stream>>>(outW, outW16, Cc*Cc);
    cvt16_kernel<<<(ICc*Cc + 255)/256, blk256, 0, stream>>>(l1W, l1W16, ICc*Cc);
    cvt16_kernel<<<(Cc*ICc + 255)/256, blk256, 0, stream>>>(l2W, l2W16, Cc*ICc);

    // Stage 1: KNN + conv1 + max_k ; gc projection
    knn_conv1_kernel<<<gridPts, blk256, 0, stream>>>(x, conv1W, conv1b, fbuf, f16);
    gc_kernel<<<gridPts, blk256, 0, stream>>>(x, g1W, g1b, g2W, g2b, gc16);

    // Stage 2: q,k,v projections (WMMA), q pre-scaled by 1/sqrt(C)
    gemm_nt_kernel<1><<<gridG, blkg, 0, stream>>>(qW16, 0, f16, sF, NP,
        nullptr, nullptr, nullptr, nullptr, nullptr, qs16, NP,
        Cc, Cc, Nn, 0.08838834764831845f);
    gemm_nt_kernel<1><<<gridG, blkg, 0, stream>>>(kW16, 0, f16, sF, NP,
        nullptr, nullptr, nullptr, nullptr, nullptr, kg16, NP, Cc, Cc, Nn, 1.0f);
    gemm_nt_kernel<1><<<gridG, blkg, 0, stream>>>(vW16, 0, f16, sF, NP,
        nullptr, nullptr, nullptr, nullptr, nullptr, v16b, NP, Cc, Cc, Nn, 1.0f);

    // kg = k + gc  (q/t . k^T + q/t . gc^T == q/t . (k+gc)^T)
    const int tot16 = Bsz * Cc * NP;
    add16_kernel<<<(tot16 + 255) / 256, blk256, 0, stream>>>(kg16, gc16, tot16);

    // Stage 3: attention Gram (reduce over n) + softmax + apply
    gemm_tn_kernel<<<gridA, blkg, 0, stream>>>(qs16, sF, kg16, sF, NP,
                                               attn32, Cc, NP, Cc);
    softmax_kernel<<<Bsz * Cc, 128, 0, stream>>>(attn32, attn16);
    // f1 = attn*v + f   (in-place residual into fbuf)
    gemm_nt_kernel<0><<<gridG, blkg, 0, stream>>>(attn16, (long long)Cc * Cc,
        v16b, sF, NP, nullptr, nullptr, fbuf, nullptr, fbuf, nullptr, 0,
        Cc, Cc, Nn, 1.0f);

    // Stage 4: LN1, ic = in_W*f_ln1 + in_b
    ln_kernel<<<gridPts, blk256, 0, stream>>>(fbuf, ln1g, ln1b, fln1, fln116);
    gemm_nt_kernel<0><<<gridG, blkg, 0, stream>>>(inW16, 0, fln116, sF, NP,
        inb, nullptr, nullptr, nullptr, ic32, ic16, NP, Cc, Cc, Nn, 1.0f);

    // Stage 5: SE branch — pooled s2+s3, per-point s1, sigmoid gate, out proj
    pool_kernel<<<dim3(Cc, Bsz), blk256, 0, stream>>>(ic32, avgb, mxb);
    s23_kernel<<<Bsz, Cc, 0, stream>>>(avgb, mxb, ga1W, ga1b, ga2W, ga2b,
                                       gm1W, gm1b, gm2W, gm2b, s23);
    gemm_nt_kernel<2><<<gridH, blkg, 0, stream>>>(l1W16, 0, ic16, sF, NP,
        l1b, nullptr, nullptr, nullptr, nullptr, hid16, NP, ICc, Cc, Nn, 1.0f);
    gemm_nt_kernel<3><<<gridG, blkg, 0, stream>>>(l2W16, 0, hid16,
        (long long)ICc * NP, NP, l2b, s23, nullptr, nullptr, nullptr, sc16, NP,
        Cc, ICc, Nn, 1.0f);
    // mb = out_W*scale + out_b + ic + f_ln1  (fbuf reused)
    gemm_nt_kernel<0><<<gridG, blkg, 0, stream>>>(outW16, 0, sc16, sF, NP,
        outb, nullptr, ic32, fln1, fbuf, nullptr, 0, Cc, Cc, Nn, 1.0f);

    // Stage 6: LN2 + logits + point weights
    zero_kernel<<<1, 32, 0, stream>>>(wsum, Bsz);
    ln2_w_kernel<<<gridPts, blk256, 0, stream>>>(fbuf, fln1, ln2g, ln2b,
                                                 finW, finb, wbuf, wsum);

    // Stage 7: weighted 9x9 Gram + Jacobi eigh -> e_hat (B x 9)
    xwx_kernel<<<Bsz, blk256, 0, stream>>>(x, wbuf, wsum, xwx);
    eigh_kernel<<<1, 32, 0, stream>>>(xwx, (float*)d_out);
}
